// multi_head_GAT_80685255622923
// MI455X (gfx1250) — compile-verified, matrix operand order checked by reading, not run on hardware
//
#include <hip/hip_runtime.h>

// ---------------------------------------------------------------------------
// Multi-head GAT forward for MI455X (gfx1250, wave32, WMMA).
//   H   = X @ W^T + b                  (fp32 WMMA 16x16x4)
//   s   = H @ a_s ; r = H @ a_r        (fused into k_hidden via LDS atomics)
//   e_ij = A_ij ? leaky(s_i + r_j) : -9e15 ; C = rowsoftmax(e)
//   out = headsum(C @ H)               (bf16 WMMA 16x16x32, f32 accumulate,
//                                       A streamed via async-to-LDS pipeline)
// ---------------------------------------------------------------------------

typedef __attribute__((ext_vector_type(2)))  float        v2f;
typedef __attribute__((ext_vector_type(8)))  float        v8f;
typedef __attribute__((ext_vector_type(16))) __bf16       v16bf;
typedef __attribute__((ext_vector_type(4)))  unsigned int v4u;

#define DEVINL __device__ __forceinline__

constexpr int   N_NODES = 8192;
constexpr int   DIN     = 256;
constexpr int   HID     = 256;   // HEADS * OUT
constexpr int   OUTD    = 64;
constexpr float ALPHA   = 0.2f;
constexpr float NEGINF  = -9e15f;

union FragBF { v16bf v; unsigned int u[8]; v4u q[2]; };

DEVINL unsigned short f2bf(float f) {
  unsigned int u = __builtin_bit_cast(unsigned int, f);
  u += 0x7FFFu + ((u >> 16) & 1u);            // round-to-nearest-even
  return (unsigned short)(u >> 16);
}
DEVINL unsigned int pack2bf(float lo, float hi) {
  return (unsigned int)f2bf(lo) | ((unsigned int)f2bf(hi) << 16);
}
DEVINL float leaky(float x) { return x > 0.0f ? x : ALPHA * x; }

// Async global->LDS copy of 8 bytes per lane (CDNA5, ASYNCcnt-tracked).
// vdst = VGPR holding wave-relative LDS byte address; vaddr = 64-bit address.
DEVINL void async_load_b64(unsigned lds_off, const void* gaddr) {
  asm volatile("global_load_async_to_lds_b64 %0, %1, off"
               :: "v"(lds_off), "v"((unsigned long long)(size_t)gaddr)
               : "memory");
}
#define WAIT_ASYNC_LE1() asm volatile("s_wait_asynccnt 0x1" ::: "memory")

// ---------------------------------------------------------------------------
// Kernel 1: H = X@W^T + b (fp32 WMMA), emit bf16 transposed HbT[n][i],
// and fused s = H@a_s, r = H@a_r per 16-row tile.
// Block = 256 threads = 8 waves; wave w owns column tiles n0 = 32w, 32w+16.
// ---------------------------------------------------------------------------
__global__ void __launch_bounds__(256)
k_hidden(const float* __restrict__ X, const float* __restrict__ W,
         const float* __restrict__ bias,
         const float* __restrict__ a_s, const float* __restrict__ a_r,
         unsigned short* __restrict__ HbT,
         float* __restrict__ s_out, float* __restrict__ r_out)
{
  __shared__ float sAcc[16], rAcc[16];
  const int tid  = threadIdx.x;
  const int wave = tid >> 5;
  const int lane = tid & 31;
  const int lidx = lane & 15;
  const int mBase = (lane < 16) ? 0 : 8;   // C/D layout: M = vgpr + 8*lanehalf
  const int kOff  = (lane < 16) ? 0 : 2;   // f32 A/B layout: K split 0/2
  const int i0 = blockIdx.x * 16;

  if (tid < 16) { sAcc[tid] = 0.0f; rAcc[tid] = 0.0f; }
  __syncthreads();

  const int n0a = wave * 32;
  const int n0b = n0a + 16;

  const float* xRow  = X + (size_t)(i0  + lidx) * DIN + kOff;
  const float* wRowA = W + (size_t)(n0a + lidx) * DIN + kOff;  // B[k][n] = W[n][k]
  const float* wRowB = W + (size_t)(n0b + lidx) * DIN + kOff;

  v8f c0 = {}; v8f c1 = {};
  for (int k = 0; k < DIN; k += 4) {
    v2f a  = *(const v2f*)(xRow  + k);
    v2f b0 = *(const v2f*)(wRowA + k);
    v2f b1 = *(const v2f*)(wRowB + k);
    c0 = __builtin_amdgcn_wmma_f32_16x16x4_f32(false, a, false, b0, (short)0, c0, false, false);
    c1 = __builtin_amdgcn_wmma_f32_16x16x4_f32(false, a, false, b1, (short)0, c1, false, false);
  }

  const float biasA = bias[n0a + lidx], biasB = bias[n0b + lidx];
  const float asA = a_s[n0a + lidx], asB = a_s[n0b + lidx];
  const float arA = a_r[n0a + lidx], arB = a_r[n0b + lidx];

  float hA[8], hB[8];
  for (int v = 0; v < 8; ++v) { hA[v] = c0[v] + biasA; hB[v] = c1[v] + biasB; }

  // bf16 transposed store: HbT[n * N + i]; 8 consecutive i per lane -> one b128
  v4u pa, pb2;
  for (int v = 0; v < 4; ++v) {
    pa[v]  = pack2bf(hA[2*v], hA[2*v+1]);
    pb2[v] = pack2bf(hB[2*v], hB[2*v+1]);
  }
  *(v4u*)(HbT + (size_t)(n0a + lidx) * N_NODES + i0 + mBase) = pa;
  *(v4u*)(HbT + (size_t)(n0b + lidx) * N_NODES + i0 + mBase) = pb2;

  // fused s/r reductions (each lane owns columns n0a+lidx, n0b+lidx)
  for (int v = 0; v < 8; ++v) {
    int row = mBase + v;
    atomicAdd(&sAcc[row], hA[v] * asA + hB[v] * asB);
    atomicAdd(&rAcc[row], hA[v] * arA + hB[v] * arB);
  }
  __syncthreads();
  if (tid < 16) { s_out[i0 + tid] = sAcc[tid]; r_out[i0 + tid] = rAcc[tid]; }
}

// ---------------------------------------------------------------------------
// Kernel 2: per-row online softmax stats (m_i, 1/Z_i). One block per row.
// Matches reference semantics incl. all-masked rows (uniform 1/N).
// ---------------------------------------------------------------------------
__global__ void __launch_bounds__(256)
k_stats(const int* __restrict__ A, const float* __restrict__ s,
        const float* __restrict__ r,
        float* __restrict__ m_out, float* __restrict__ zinv_out)
{
  __shared__ float smM[256], smZ[256];
  const int i = blockIdx.x;
  const int t = threadIdx.x;
  const float si = s[i];
  const int* row = A + (size_t)i * N_NODES;

  float m = NEGINF, z = 0.0f;
  for (int j = t; j < N_NODES; j += 256) {
    float val = row[j] ? leaky(si + r[j]) : NEGINF;
    if (val > m) { z = z * __expf(m - val) + 1.0f; m = val; }
    else         { z += __expf(val - m); }
  }
  smM[t] = m; smZ[t] = z;
  __syncthreads();
  for (int off = 128; off > 0; off >>= 1) {
    if (t < off) {
      float m1 = smM[t], z1 = smZ[t];
      float m2 = smM[t + off], z2 = smZ[t + off];
      float mm = fmaxf(m1, m2);
      smM[t] = mm;
      smZ[t] = z1 * __expf(m1 - mm) + z2 * __expf(m2 - mm);
    }
    __syncthreads();
  }
  if (t == 0) { m_out[i] = smM[0]; zinv_out[i] = 1.0f / smZ[0]; }
}

// ---------------------------------------------------------------------------
// Kernel 3: flash-style fused softmax-weighted GEMM + head sum.
// Block = 256 threads = 8 waves, 16 output rows, full 256 hid cols.
// A tile (the only HBM-resident operand) is double-buffered one tile ahead
// via global_load_async_to_lds_b64 / s_wait_asynccnt. Per j-step of 32:
// 16x32 bf16 prob tile built once in LDS (A-fragment order), two
// v_wmma_f32_16x16x32_bf16 per wave against L2-resident HbT (b128 gathers).
// ---------------------------------------------------------------------------
__global__ void __launch_bounds__(256)
k_attn(const int* __restrict__ A, const float* __restrict__ s,
       const float* __restrict__ r, const float* __restrict__ m_arr,
       const float* __restrict__ zinv, const unsigned short* __restrict__ HbT,
       float* __restrict__ out)
{
  __shared__ __align__(16) unsigned int pAu[256];  // 16x32 bf16 prob tile
  __shared__ __align__(8)  int aBuf[2][512];       // double-buffered A staging
  __shared__ float outS[16 * 64];
  const int tid  = threadIdx.x;
  const int wave = tid >> 5;
  const int lane = tid & 31;
  const int lidx = lane & 15;
  const int mBase = (lane < 16) ? 0 : 8;
  const int i0 = blockIdx.x * 16;

  for (int q = tid; q < 16 * 64; q += 256) outS[q] = 0.0f;

  // probability-tile writer mapping: this thread produces the K-pair of
  // fragment "vgpr" wv for fragment lane wl (A 16x32 bf16 layout).
  const int wl = tid >> 3;                       // fragment lane 0..31
  const int wv = tid & 7;                        // fragment vgpr 0..7
  const int wM = wl & 15;                        // matrix row within tile
  const int wK = ((wl < 16) ? 0 : 8) + ((wv < 4) ? 2 * wv : 16 + 2 * (wv - 4));
  const float sv = s[i0 + wM];
  const float mv = m_arr[i0 + wM];
  const int* aRow = A + (size_t)(i0 + wM) * N_NODES + wK;

  // B-operand pointers: HbT is [hid][N] bf16; per lane the 8 fragment uints
  // are two contiguous runs of 4 -> two global_load_b128 per fragment.
  const unsigned int* Hb32 = (const unsigned int*)HbT;
  const int n0a = wave * 32;
  const int n0b = n0a + 16;
  const int kuBase = (lane < 16) ? 0 : 4;        // (K lane-half split)/2
  const unsigned int* hbA = Hb32 + (size_t)(n0a + lidx) * (N_NODES / 2) + kuBase;
  const unsigned int* hbB = Hb32 + (size_t)(n0b + lidx) * (N_NODES / 2) + kuBase;

  v8f c0 = {}; v8f c1 = {};

  // prologue: async-stage tile 0
  async_load_b64((unsigned)(size_t)&aBuf[0][2 * tid], aRow);

  for (int j0 = 0; j0 < N_NODES; j0 += 32) {
    const int buf = (j0 >> 5) & 1;
    // async-stage next tile (wraps on last iter; keeps wait-count invariant)
    const int jn = (j0 + 32) & (N_NODES - 1);
    async_load_b64((unsigned)(size_t)&aBuf[buf ^ 1][2 * tid], aRow + jn);
    WAIT_ASYNC_LE1();                            // current tile has landed
    __syncthreads();
    {
      int2 a2 = *(const int2*)&aBuf[buf][2 * tid];
      v2f rr = *(const v2f*)(r + j0 + wK);
      float v0 = a2.x ? leaky(sv + rr.x) : NEGINF;
      float v1 = a2.y ? leaky(sv + rr.y) : NEGINF;
      pAu[tid] = pack2bf(__expf(v0 - mv), __expf(v1 - mv));
    }
    __syncthreads();

    FragBF af, b0, b1;
    const v4u* pls = (const v4u*)&pAu[lane * 8];
    af.q[0] = pls[0];
    af.q[1] = pls[1];
    const int ku = j0 >> 1;
    b0.q[0] = *(const v4u*)(hbA + ku);
    b0.q[1] = *(const v4u*)(hbA + ku + 8);
    b1.q[0] = *(const v4u*)(hbB + ku);
    b1.q[1] = *(const v4u*)(hbB + ku + 8);
    c0 = __builtin_amdgcn_wmma_f32_16x16x32_bf16(false, af.v, false, b0.v, (short)0, c0, false, false);
    c1 = __builtin_amdgcn_wmma_f32_16x16x32_bf16(false, af.v, false, b1.v, (short)0, c1, false, false);
  }

  __syncthreads();
  // scale by 1/Z and head-sum (hid col n -> output col n & 63)
  for (int v = 0; v < 8; ++v) {
    int M = mBase + v;
    float sc = zinv[i0 + M];
    atomicAdd(&outS[M * 64 + ((n0a + lidx) & 63)], c0[v] * sc);
    atomicAdd(&outS[M * 64 + ((n0b + lidx) & 63)], c1[v] * sc);
  }
  __syncthreads();
  for (int q = tid; q < 16 * 64; q += 256) {
    int row = q >> 6, o = q & 63;
    out[(size_t)(i0 + row) * OUTD + o] = outS[q];
  }
}

// ---------------------------------------------------------------------------
extern "C" void kernel_launch(void* const* d_in, const int* in_sizes, int n_in,
                              void* d_out, int out_size, void* d_ws, size_t ws_size,
                              hipStream_t stream) {
  (void)in_sizes; (void)n_in; (void)out_size; (void)ws_size;
  const float* X   = (const float*)d_in[0];
  const int*   A   = (const int*)  d_in[1];
  const float* W   = (const float*)d_in[2];
  const float* bia = (const float*)d_in[3];
  const float* a_s = (const float*)d_in[4];
  const float* a_r = (const float*)d_in[5];
  float* out = (float*)d_out;

  // workspace: HbT (bf16, 4MB) + s, r, m, 1/Z (f32, 32KB each)
  unsigned short* HbT = (unsigned short*)d_ws;
  float* fbase = (float*)((char*)d_ws + (size_t)N_NODES * HID * sizeof(unsigned short));
  float* s    = fbase;
  float* r    = fbase + N_NODES;
  float* mrow = fbase + 2 * N_NODES;
  float* zinv = fbase + 3 * N_NODES;

  k_hidden<<<N_NODES / 16, 256, 0, stream>>>(X, W, bia, a_s, a_r, HbT, s, r);
  k_stats <<<N_NODES,      256, 0, stream>>>(A, s, r, mrow, zinv);
  k_attn  <<<N_NODES / 16, 256, 0, stream>>>(A, s, r, mrow, zinv, HbT, out);
}
